// Encoder_Cross_46385646797321
// MI455X (gfx1250) — compile-verified, matrix operand order checked by reading, not run on hardware
//
#include <hip/hip_runtime.h>
#include <cmath>

// ---------------------------------------------------------------------------
// Types for CDNA5 WMMA (wave32, v_wmma_f32_16x16x32_bf16)
// ---------------------------------------------------------------------------
typedef __bf16 bf16;
typedef __attribute__((ext_vector_type(16))) __bf16 v16bf;
typedef __attribute__((ext_vector_type(8)))  __bf16 v8bf;
typedef __attribute__((ext_vector_type(8)))  float  v8f;

union FragU { v16bf v; v8bf h[2]; };

#define WMMA_BF16(a, b, c) \
  __builtin_amdgcn_wmma_f32_16x16x32_bf16(false, (a), false, (b), (short)0, (c), false, false)

__device__ inline v8bf zero8() {
  v8bf z;
#pragma unroll
  for (int i = 0; i < 8; ++i) z[i] = (__bf16)0.0f;
  return z;
}

// Fragment load for a [rows, K] row-major bf16 matrix.
// ISA 16-bit A/B layout: lanes 0-15 hold row=rowBase+lane, K = kk..kk+7 (v0-3)
// and kk+16..kk+23 (v4-7); lanes 16-31 hold the same rows, K offset +8.
__device__ inline v16bf frag_load(const bf16* __restrict__ M, size_t ld,
                                  int rowBase, int kk, int lane) {
  int r  = rowBase + (lane & 15);
  int k0 = kk + ((lane >> 4) << 3);
  const bf16* p = M + (size_t)r * ld + k0;
  FragU u;
  u.h[0] = *(const v8bf*)(p);
  u.h[1] = *(const v8bf*)(p + 16);
  return u.v;
}

// Guarded version (zero-fills K tail; K is a multiple of 8).
__device__ inline v16bf frag_load_g(const bf16* __restrict__ M, size_t ld,
                                    int rowBase, int kk, int lane, int K) {
  int r  = rowBase + (lane & 15);
  int k0 = kk + ((lane >> 4) << 3);
  const bf16* p = M + (size_t)r * ld;
  FragU u;
  u.h[0] = (k0 + 8  <= K) ? *(const v8bf*)(p + k0)      : zero8();
  u.h[1] = (k0 + 24 <= K) ? *(const v8bf*)(p + k0 + 16) : zero8();
  return u.v;
}

// C/D 16x16 f32 layout: VGPR r, lanes 0-15 -> (M=r, N=lane); lanes 16-31 -> (M=8+r).
template <bool OUT_BF16, bool BIAS, bool GELU>
__device__ inline void frag_store(v8f c, void* __restrict__ Cv, size_t ldc,
                                  int rowBase, int colBase, int lane,
                                  const float* __restrict__ bias) {
  int col = colBase + (lane & 15);
  float bv = 0.0f;
  if (BIAS) bv = bias[col];
  int r0 = rowBase + ((lane >> 4) << 3);
#pragma unroll
  for (int r = 0; r < 8; ++r) {
    float val = c[r] + bv;
    if (GELU) val = 0.5f * val * (1.0f + erff(val * 0.70710678118654752f));
    size_t idx = (size_t)(r0 + r) * ldc + col;
    if (OUT_BF16) ((bf16*)Cv)[idx] = (bf16)val;
    else          ((float*)Cv)[idx] = val;
  }
}

// ---------------------------------------------------------------------------
// Generic strided-batched GEMM:  C[bp] = A[bp] (MxK) * B[bp] (stored [N,K]) + bias
// Batch base (elements) = (bp/div)*hi + (bp%div)*lo  for each of A, B, C.
// ---------------------------------------------------------------------------
struct GemmP {
  const bf16* A; const bf16* B; void* C; const float* bias;
  int M, N, K;
  size_t lda, ldb, ldc;
  int divA; size_t hiA, loA;
  int divB; size_t hiB, loB;
  int divC; size_t hiC, loC;
};

// Big tiles: block = 128 thr (4 waves, 2x2), block tile 128x128, wave tile
// 64x64 (16 accumulator fragments -> 16 WMMAs per 8 fragment loads per K=32).
// All call sites use M % 128 == 0 (or M == 128) and N % 128 == 0, K % 32 == 0.
template <bool OUT_BF16, bool BIAS, bool GELU>
__global__ void __launch_bounds__(128, 1) gemm_big(GemmP p) {
  int bp = blockIdx.z;
  const bf16* A = p.A + (size_t)(bp / p.divA) * p.hiA + (size_t)(bp % p.divA) * p.loA;
  const bf16* B = p.B + (size_t)(bp / p.divB) * p.hiB + (size_t)(bp % p.divB) * p.loB;
  size_t cOff = (size_t)(bp / p.divC) * p.hiC + (size_t)(bp % p.divC) * p.loC;
  void* C = OUT_BF16 ? (void*)((bf16*)p.C + cOff) : (void*)((float*)p.C + cOff);

  int tid = threadIdx.x, wave = tid >> 5, lane = tid & 31;
  int rowBase = blockIdx.y * 128 + (wave >> 1) * 64;
  int colBase = blockIdx.x * 128 + (wave & 1) * 64;

  v8f acc[4][4];
#pragma unroll
  for (int i = 0; i < 4; ++i)
#pragma unroll
    for (int j = 0; j < 4; ++j)
#pragma unroll
      for (int e = 0; e < 8; ++e) acc[i][j][e] = 0.f;

  for (int kk = 0; kk < p.K; kk += 32) {
    if (kk + 32 < p.K) {  // prefetch next K slab (global_prefetch_b8)
      __builtin_prefetch(A + (size_t)(rowBase + lane)      * p.lda + kk + 32, 0, 1);
      __builtin_prefetch(A + (size_t)(rowBase + 32 + lane) * p.lda + kk + 32, 0, 1);
      __builtin_prefetch(B + (size_t)(colBase + lane)      * p.ldb + kk + 32, 0, 1);
      __builtin_prefetch(B + (size_t)(colBase + 32 + lane) * p.ldb + kk + 32, 0, 1);
    }
    v16bf b0 = frag_load(B, p.ldb, colBase,      kk, lane);
    v16bf b1 = frag_load(B, p.ldb, colBase + 16, kk, lane);
    v16bf b2 = frag_load(B, p.ldb, colBase + 32, kk, lane);
    v16bf b3 = frag_load(B, p.ldb, colBase + 48, kk, lane);
#pragma unroll
    for (int i = 0; i < 4; ++i) {
      v16bf a = frag_load(A, p.lda, rowBase + i * 16, kk, lane);
      acc[i][0] = WMMA_BF16(a, b0, acc[i][0]);
      acc[i][1] = WMMA_BF16(a, b1, acc[i][1]);
      acc[i][2] = WMMA_BF16(a, b2, acc[i][2]);
      acc[i][3] = WMMA_BF16(a, b3, acc[i][3]);
    }
  }
#pragma unroll
  for (int i = 0; i < 4; ++i)
#pragma unroll
    for (int j = 0; j < 4; ++j)
      frag_store<OUT_BF16, BIAS, GELU>(acc[i][j], C, p.ldc,
                                       rowBase + i * 16, colBase + j * 16,
                                       lane, p.bias);
}

// M == 16 kernel (attn2 score / AV GEMMs); one 16x16 tile per wave; guarded K.
template <bool OUT_BF16>
__global__ void __launch_bounds__(128) gemm_m16(GemmP p) {
  int bp = blockIdx.z;
  const bf16* A = p.A + (size_t)(bp / p.divA) * p.hiA + (size_t)(bp % p.divA) * p.loA;
  const bf16* B = p.B + (size_t)(bp / p.divB) * p.hiB + (size_t)(bp % p.divB) * p.loB;
  size_t cOff = (size_t)(bp / p.divC) * p.hiC + (size_t)(bp % p.divC) * p.loC;
  void* C = OUT_BF16 ? (void*)((bf16*)p.C + cOff) : (void*)((float*)p.C + cOff);

  int wave = threadIdx.x >> 5, lane = threadIdx.x & 31;
  int colBase = (blockIdx.x * 4 + wave) * 16;
  if (colBase >= p.N) return;

  v8f c;
#pragma unroll
  for (int i = 0; i < 8; ++i) c[i] = 0.f;

  for (int kk = 0; kk < p.K; kk += 32) {
    v16bf a = frag_load_g(A, p.lda, 0,       kk, lane, p.K);
    v16bf b = frag_load_g(B, p.ldb, colBase, kk, lane, p.K);
    c = WMMA_BF16(a, b, c);
  }
  frag_store<OUT_BF16, false, false>(c, C, p.ldc, 0, colBase, lane, nullptr);
}

// ---------------------------------------------------------------------------
// Data-movement / elementwise kernels
// ---------------------------------------------------------------------------

// W[K,N] f32  ->  Wt[N,K] bf16 (LDS tiled transpose)
__global__ void __launch_bounds__(256) transpose_f32_bf16(const float* __restrict__ src,
                                                          bf16* __restrict__ dst,
                                                          int K, int N) {
  __shared__ float tile[32][33];
  int nBase = blockIdx.x * 32, kBase = blockIdx.y * 32;
  int tx = threadIdx.x, ty = threadIdx.y;
#pragma unroll
  for (int i = 0; i < 4; ++i)
    tile[ty + i * 8][tx] = src[(size_t)(kBase + ty + i * 8) * N + nBase + tx];
  __syncthreads();
#pragma unroll
  for (int i = 0; i < 4; ++i)
    dst[(size_t)(nBase + ty + i * 8) * K + kBase + tx] = (bf16)tile[tx][ty + i * 8];
}

__global__ void f32_to_bf16(const float* __restrict__ s, bf16* __restrict__ d, long long n) {
  long long i = (long long)blockIdx.x * blockDim.x + threadIdx.x;
  if (i < n) d[i] = (bf16)s[i];
}

// dst[bp][d*S + s] = src[row(bp,s)*srcLd + d], row = (bp/div)*hi + (bp%div)*lo + s*sStride
__global__ void batched_transpose_bf16(const bf16* __restrict__ src, bf16* __restrict__ dst,
                                       int D, int S, int divb, size_t hi, size_t lo,
                                       size_t sStride, size_t srcLd, long long total) {
  long long idx = (long long)blockIdx.x * blockDim.x + threadIdx.x;
  if (idx >= total) return;
  long long per = (long long)D * S;
  int bp = (int)(idx / per);
  int rem = (int)(idx % per);
  int d = rem / S, s = rem % S;
  size_t row = (size_t)(bp / divb) * hi + (size_t)(bp % divb) * lo + (size_t)s * sStride;
  dst[idx] = src[row * srcLd + d];
}

// attn1 softmax: rows of 128, scaled, unmasked; bf16 out.
__global__ void __launch_bounds__(128) softmax_p(const float* __restrict__ S,
                                                 bf16* __restrict__ Sb, float scale) {
  int row = blockIdx.x, t = threadIdx.x;
  __shared__ float red[128];
  float v = S[(size_t)row * 128 + t] * scale;
  red[t] = v; __syncthreads();
  for (int o = 64; o > 0; o >>= 1) { if (t < o) red[t] = fmaxf(red[t], red[t + o]); __syncthreads(); }
  float m = red[0]; __syncthreads();
  float e = __expf(v - m);
  red[t] = e; __syncthreads();
  for (int o = 64; o > 0; o >>= 1) { if (t < o) red[t] += red[t + o]; __syncthreads(); }
  Sb[(size_t)row * 128 + t] = (bf16)(e / red[0]);
}

// attn2 softmax: one thread per (b,p,v) row of 16; masked by var_mask[b,v,r].
__global__ void softmax_v(const float* __restrict__ S, bf16* __restrict__ Sb,
                          const unsigned char* __restrict__ mask,
                          int Pc, int Vc, long long rows, float scale) {
  long long row = (long long)blockIdx.x * blockDim.x + threadIdx.x;
  if (row >= rows) return;
  int v = (int)(row % Vc);
  long long bp = row / Vc;
  int b = (int)(bp / Pc);
  const float* s = S + row * Vc;
  const unsigned char* mk = mask + ((size_t)b * Vc + v) * Vc;
  float m = -1e30f;
#pragma unroll
  for (int r = 0; r < 16; ++r) if (!mk[r]) m = fmaxf(m, s[r] * scale);
  float e[16]; float sum = 0.f;
#pragma unroll
  for (int r = 0; r < 16; ++r) { e[r] = mk[r] ? 0.f : __expf(s[r] * scale - m); sum += e[r]; }
  float inv = 1.0f / sum;
#pragma unroll
  for (int r = 0; r < 16; ++r) Sb[row * Vc + r] = (bf16)(e[r] * inv);
}

// y = LN(resid + delta); write f32 stream + bf16 copy. One block per row (PD=2048).
__global__ void __launch_bounds__(256) ln_resid(const float* __restrict__ resid,
                                                const float* __restrict__ delta,
                                                const float* __restrict__ g,
                                                const float* __restrict__ be,
                                                float* __restrict__ Hout,
                                                bf16* __restrict__ Xb) {
  int row = blockIdx.x, t = threadIdx.x;
  size_t base = (size_t)row * 2048;
  float v[8]; float s = 0.f;
#pragma unroll
  for (int i = 0; i < 8; ++i) { int c = t + i * 256; v[i] = resid[base + c] + delta[base + c]; s += v[i]; }
  __shared__ float red[256];
  red[t] = s; __syncthreads();
  for (int o = 128; o > 0; o >>= 1) { if (t < o) red[t] += red[t + o]; __syncthreads(); }
  float mu = red[0] * (1.0f / 2048.0f);
  __syncthreads();
  s = 0.f;
#pragma unroll
  for (int i = 0; i < 8; ++i) { float d = v[i] - mu; s += d * d; }
  red[t] = s; __syncthreads();
  for (int o = 128; o > 0; o >>= 1) { if (t < o) red[t] += red[t + o]; __syncthreads(); }
  float rstd = rsqrtf(red[0] * (1.0f / 2048.0f) + 1e-5f);
#pragma unroll
  for (int i = 0; i < 8; ++i) {
    int c = t + i * 256;
    float y = (v[i] - mu) * rstd * g[c] + be[c];
    Hout[base + c] = y; Xb[base + c] = (bf16)y;
  }
}

// Final LN -> d_out[x_out] and d_out[x_next] (x_next is a pure reshape alias).
__global__ void __launch_bounds__(256) ln_final(const float* __restrict__ resid,
                                                const float* __restrict__ delta,
                                                const float* __restrict__ g,
                                                const float* __restrict__ be,
                                                float* __restrict__ out, size_t dup) {
  int row = blockIdx.x, t = threadIdx.x;
  size_t base = (size_t)row * 2048;
  float v[8]; float s = 0.f;
#pragma unroll
  for (int i = 0; i < 8; ++i) { int c = t + i * 256; v[i] = resid[base + c] + delta[base + c]; s += v[i]; }
  __shared__ float red[256];
  red[t] = s; __syncthreads();
  for (int o = 128; o > 0; o >>= 1) { if (t < o) red[t] += red[t + o]; __syncthreads(); }
  float mu = red[0] * (1.0f / 2048.0f);
  __syncthreads();
  s = 0.f;
#pragma unroll
  for (int i = 0; i < 8; ++i) { float d = v[i] - mu; s += d * d; }
  red[t] = s; __syncthreads();
  for (int o = 128; o > 0; o >>= 1) { if (t < o) red[t] += red[t + o]; __syncthreads(); }
  float rstd = rsqrtf(red[0] * (1.0f / 2048.0f) + 1e-5f);
#pragma unroll
  for (int i = 0; i < 8; ++i) {
    int c = t + i * 256;
    float y = (v[i] - mu) * rstd * g[c] + be[c];
    out[base + c] = y; out[dup + base + c] = y;
  }
}

// ---------------------------------------------------------------------------
// Host side
// ---------------------------------------------------------------------------
static inline GemmP mkGemm(const bf16* A, const bf16* B, void* C, const float* bias,
                           int M, int N, int K, size_t lda, size_t ldb, size_t ldc) {
  GemmP p{};
  p.A = A; p.B = B; p.C = C; p.bias = bias;
  p.M = M; p.N = N; p.K = K; p.lda = lda; p.ldb = ldb; p.ldc = ldc;
  p.divA = p.divB = p.divC = 1;
  p.hiA = p.loA = p.hiB = p.loB = p.hiC = p.loC = 0;
  return p;
}

extern "C" void kernel_launch(void* const* d_in, const int* in_sizes, int n_in,
                              void* d_out, int out_size, void* d_ws, size_t ws_size,
                              hipStream_t stream) {
  (void)in_sizes; (void)n_in; (void)out_size; (void)ws_size;
  const int Bc = 8, Vc = 16, Pc = 128;
  const int PD = 2048, HD = 8192;
  const size_t MR  = (size_t)Bc * Vc * Pc;     // 16384 rows
  const size_t TOT = MR * PD;                  // 33,554,432 elements
  const float scale = 0.022097086912079608f;   // 1/sqrt(2048)

  // ---- inputs ----
  const float* x    = (const float*)d_in[0];
  const unsigned char* vmask = (const unsigned char*)d_in[1];
  const float *wq1 = (const float*)d_in[2],  *bq1 = (const float*)d_in[3];
  const float *wk1 = (const float*)d_in[4],  *bk1 = (const float*)d_in[5];
  const float *wo1 = (const float*)d_in[6],  *bo1 = (const float*)d_in[7];
  const float *wq2 = (const float*)d_in[8],  *bq2 = (const float*)d_in[9];
  const float *wk2 = (const float*)d_in[10], *bk2 = (const float*)d_in[11];
  const float *wo2 = (const float*)d_in[12], *bo2 = (const float*)d_in[13];
  const float *g1 = (const float*)d_in[14], *be1 = (const float*)d_in[15];
  const float *g2 = (const float*)d_in[16], *be2 = (const float*)d_in[17];
  const float *g3 = (const float*)d_in[18], *be3 = (const float*)d_in[19];
  const float *wl1 = (const float*)d_in[20], *bl1 = (const float*)d_in[21];
  const float *wl2 = (const float*)d_in[22], *bl2 = (const float*)d_in[23];
  float* out = (float*)d_out;

  // ---- workspace carve ----
  char* ws = (char*)d_ws; size_t off = 0;
  auto carve = [&](size_t bytes) { void* p = ws + off; off = (off + bytes + 255) & ~(size_t)255; return p; };
  bf16* WQ1T = (bf16*)carve((size_t)PD * PD * 2);
  bf16* WK1T = (bf16*)carve((size_t)PD * PD * 2);
  bf16* WO1T = (bf16*)carve((size_t)PD * PD * 2);
  bf16* WQ2T = (bf16*)carve((size_t)PD * PD * 2);
  bf16* WK2T = (bf16*)carve((size_t)PD * PD * 2);
  bf16* WO2T = (bf16*)carve((size_t)PD * PD * 2);
  bf16* WL1T = (bf16*)carve((size_t)PD * HD * 2);   // [8192,2048]
  bf16* WL2T = (bf16*)carve((size_t)PD * HD * 2);   // [2048,8192]
  bf16* XBF  = (bf16*)carve(TOT * 2);
  bf16* Qb   = (bf16*)carve(TOT * 2);
  bf16* KVb  = (bf16*)carve(TOT * 2);
  bf16* KVT  = (bf16*)carve(TOT * 2);
  bf16* A1   = (bf16*)carve(TOT * 2);
  bf16* Zb   = (bf16*)carve(MR * (size_t)HD * 2);
  bf16* SBF  = (bf16*)carve((size_t)Bc * Vc * Pc * Pc * 2);
  float* Sf  = (float*)carve((size_t)Bc * Vc * Pc * Pc * 4);
  float* AO  = (float*)carve(TOT * 4);
  float* H   = (float*)carve(TOT * 4);

  dim3 tb(32, 8);
  // ---- weight conversion (transposed bf16 [N,K]) ----
  transpose_f32_bf16<<<dim3(PD / 32, PD / 32), tb, 0, stream>>>(wq1, WQ1T, PD, PD);
  transpose_f32_bf16<<<dim3(PD / 32, PD / 32), tb, 0, stream>>>(wk1, WK1T, PD, PD);
  transpose_f32_bf16<<<dim3(PD / 32, PD / 32), tb, 0, stream>>>(wo1, WO1T, PD, PD);
  transpose_f32_bf16<<<dim3(PD / 32, PD / 32), tb, 0, stream>>>(wq2, WQ2T, PD, PD);
  transpose_f32_bf16<<<dim3(PD / 32, PD / 32), tb, 0, stream>>>(wk2, WK2T, PD, PD);
  transpose_f32_bf16<<<dim3(PD / 32, PD / 32), tb, 0, stream>>>(wo2, WO2T, PD, PD);
  transpose_f32_bf16<<<dim3(HD / 32, PD / 32), tb, 0, stream>>>(wl1, WL1T, PD, HD);
  transpose_f32_bf16<<<dim3(PD / 32, HD / 32), tb, 0, stream>>>(wl2, WL2T, HD, PD);

  f32_to_bf16<<<(int)((TOT + 255) / 256), 256, 0, stream>>>(x, XBF, (long long)TOT);

  // ================= Attn_PatchLevel =================
  { GemmP p = mkGemm(XBF, WQ1T, Qb, bq1, (int)MR, PD, PD, PD, PD, PD);
    gemm_big<true, true, false><<<dim3(PD / 128, MR / 128, 1), 128, 0, stream>>>(p); }
  { GemmP p = mkGemm(XBF, WK1T, KVb, bk1, (int)MR, PD, PD, PD, PD, PD);
    gemm_big<true, true, false><<<dim3(PD / 128, MR / 128, 1), 128, 0, stream>>>(p); }

  // scores[q] = Q[q] (128x2048) * KV[q]^T : batched over B*V=128
  { GemmP p = mkGemm(Qb, KVb, Sf, nullptr, Pc, Pc, PD, PD, PD, Pc);
    p.hiA = (size_t)Pc * PD; p.hiB = (size_t)Pc * PD; p.hiC = (size_t)Pc * Pc;
    gemm_big<false, false, false><<<dim3(1, 1, Bc * Vc), 128, 0, stream>>>(p); }

  softmax_p<<<(int)(Bc * Vc * Pc), 128, 0, stream>>>(Sf, SBF, scale);

  // KVT[q] = KV[q]^T : [2048,128] per batch (row(bp,s)=bp*128+s)
  batched_transpose_bf16<<<(int)((TOT + 255) / 256), 256, 0, stream>>>(
      KVb, KVT, PD, Pc, 1, (size_t)Pc, 0, 1, PD, (long long)TOT);

  // A1[q] = attn[q] (128x128) * KVT[q] (2048x128-as-[N,K])
  { GemmP p = mkGemm(SBF, KVT, A1, nullptr, Pc, PD, Pc, Pc, Pc, PD);
    p.hiA = (size_t)Pc * Pc; p.hiB = (size_t)PD * Pc; p.hiC = (size_t)Pc * PD;
    gemm_big<true, false, false><<<dim3(PD / 128, 1, Bc * Vc), 128, 0, stream>>>(p); }

  { GemmP p = mkGemm(A1, WO1T, AO, bo1, (int)MR, PD, PD, PD, PD, PD);
    gemm_big<false, true, false><<<dim3(PD / 128, MR / 128, 1), 128, 0, stream>>>(p); }

  ln_resid<<<(int)MR, 256, 0, stream>>>(x, AO, g1, be1, H, XBF);

  // ================= Attn_VarLevel (strided batches, no physical transpose) ==
  { GemmP p = mkGemm(XBF, WQ2T, Qb, bq2, (int)MR, PD, PD, PD, PD, PD);
    gemm_big<true, true, false><<<dim3(PD / 128, MR / 128, 1), 128, 0, stream>>>(p); }
  { GemmP p = mkGemm(XBF, WK2T, KVb, bk2, (int)MR, PD, PD, PD, PD, PD);
    gemm_big<true, true, false><<<dim3(PD / 128, MR / 128, 1), 128, 0, stream>>>(p); }

  // scores2[bp] (16x16): rows v of batch (b,p) live at row b*V*P + v*P + p
  const size_t HIbp = (size_t)Vc * Pc * PD;  // b stride (elements)
  { GemmP p = mkGemm(Qb, KVb, Sf, nullptr, Vc, Vc, PD, (size_t)Pc * PD, (size_t)Pc * PD, Vc);
    p.divA = Pc; p.hiA = HIbp; p.loA = PD;
    p.divB = Pc; p.hiB = HIbp; p.loB = PD;
    p.hiC = (size_t)Vc * Vc;
    gemm_m16<false><<<dim3(1, 1, Bc * Pc), 128, 0, stream>>>(p); }

  softmax_v<<<(int)((Bc * Pc * Vc + 255) / 256), 256, 0, stream>>>(
      Sf, SBF, vmask, Pc, Vc, (long long)Bc * Pc * Vc, scale);

  // KV2T[bp] = KV rows (strided) transposed: [2048,16] per (b,p) batch
  batched_transpose_bf16<<<(int)((TOT + 255) / 256), 256, 0, stream>>>(
      KVb, KVT, PD, Vc, Pc, (size_t)Vc * Pc, 1, Pc, PD, (long long)TOT);

  // A2[bp] = attn2 (16x16) * KV2T (2048x16-as-[N,K]) -> back into strided layout
  { GemmP p = mkGemm(SBF, KVT, A1, nullptr, Vc, PD, Vc, Vc, Vc, (size_t)Pc * PD);
    p.hiA = (size_t)Vc * Vc;
    p.hiB = (size_t)PD * Vc;
    p.divC = Pc; p.hiC = HIbp; p.loC = PD;
    gemm_m16<true><<<dim3(PD / 64, 1, Bc * Pc), 128, 0, stream>>>(p); }

  { GemmP p = mkGemm(A1, WO2T, AO, bo2, (int)MR, PD, PD, PD, PD, PD);
    gemm_big<false, true, false><<<dim3(PD / 128, MR / 128, 1), 128, 0, stream>>>(p); }

  ln_resid<<<(int)MR, 256, 0, stream>>>(H, AO, g2, be2, H, XBF);

  // ================= FFN (GELU fused into FFN1 epilogue) =================
  { GemmP p = mkGemm(XBF, WL1T, Zb, bl1, (int)MR, HD, PD, PD, PD, HD);
    gemm_big<true, true, true><<<dim3(HD / 128, MR / 128, 1), 128, 0, stream>>>(p); }
  { GemmP p = mkGemm(Zb, WL2T, AO, bl2, (int)MR, PD, HD, HD, HD, PD);
    gemm_big<false, true, false><<<dim3(PD / 128, MR / 128, 1), 128, 0, stream>>>(p); }

  // final LN -> x_out and x_next (same flat data)
  ln_final<<<(int)MR, 256, 0, stream>>>(H, AO, g3, be3, out, TOT);
}
